// CfCBlock_8813272891917
// MI455X (gfx1250) — compile-verified, matrix operand order checked by reading, not run on hardware
//
#include <hip/hip_runtime.h>

// ---------------------------------------------------------------------------
// CfC block for MI455X (gfx1250, wave32, WMMA + async-to-LDS).
//   D = H = 1024, B = 16, S = 2048.
// ---------------------------------------------------------------------------

typedef _Float16 v16h  __attribute__((ext_vector_type(16)));
typedef _Float16 f16x8 __attribute__((ext_vector_type(8)));
typedef float    v8f   __attribute__((ext_vector_type(8)));
typedef int      i32x4 __attribute__((vector_size(16)));  // matches builtin proto

#define D_MODEL 1024
#define HGRU    1024
#define BATCH   16
#define SEQ     2048
#define NROWS   (BATCH * SEQ)   // 32768 token rows
#define G3H     (3 * HGRU)      // 3072 gate rows

// ---------------------------------------------------------------------------
// Async global->LDS staging (ASYNCcnt path). Falls back to synchronous
// vector copies if the toolchain lacks the gfx1250 builtins.
// Builtin proto (from hipcc diagnostic): (v4i AS1*, v4i AS3*, imm, imm).
// ---------------------------------------------------------------------------
#if defined(__has_builtin)
#if __has_builtin(__builtin_amdgcn_global_load_async_to_lds_b128) && \
    __has_builtin(__builtin_amdgcn_s_wait_asynccnt)
#define HAS_ASYNC_LDS 1
#endif
#endif
#ifndef HAS_ASYNC_LDS
#define HAS_ASYNC_LDS 0
#endif

static __device__ __forceinline__ void cp_b128(void* lds_dst, const void* gsrc) {
#if HAS_ASYNC_LDS
  __builtin_amdgcn_global_load_async_to_lds_b128(
      (__attribute__((address_space(1))) i32x4*)gsrc,
      (__attribute__((address_space(3))) i32x4*)lds_dst, 0, 0);
#else
  *(uint4*)lds_dst = *(const uint4*)gsrc;
#endif
}

static __device__ __forceinline__ void cp_wait() {
#if HAS_ASYNC_LDS
  __builtin_amdgcn_s_wait_asynccnt(0);
#endif
}

// ---------------------------------------------------------------------------
// WMMA 16x16x32 f16 fragment loader.
// A-matrix per-lane layout (ISA 7.12.2, 16-bit A 16x32):
//   lanes 0-15  : row M = lane,    K = {k0+0..7,  k0+16..23}
//   lanes 16-31 : row M = lane-16, K = {k0+8..15, k0+24..31}
// B mirrors this with N as the striped dim when the weight is row-major
// [N, K] (PyTorch layout). Two contiguous 16-byte loads per lane.
// ---------------------------------------------------------------------------
static __device__ __forceinline__ v16h
load_frag(const _Float16* __restrict__ base, int stride, int row, int kHalf, int k0) {
  const _Float16* p = base + (size_t)row * stride + k0 + kHalf * 8;
  f16x8 lo = *(const f16x8*)(p);
  f16x8 hi = *(const f16x8*)(p + 16);
  return __builtin_shufflevector(lo, hi, 0,1,2,3,4,5,6,7,8,9,10,11,12,13,14,15);
}

// ---------------------------------------------------------------------------
// Kernel 1: LayerNorm (f32 in) -> f16 normalized activations.
// ---------------------------------------------------------------------------
__global__ __launch_bounds__(256) void ln_kernel(
    const float* __restrict__ x, const float* __restrict__ g,
    const float* __restrict__ b, _Float16* __restrict__ xn) {
  const int row = blockIdx.x;
  const int tid = threadIdx.x;
  const float* xr = x + (size_t)row * D_MODEL;
  float v[4];
  float s = 0.f, ss = 0.f;
#pragma unroll
  for (int i = 0; i < 4; ++i) {
    v[i] = xr[tid + i * 256];
    s += v[i];
    ss += v[i] * v[i];
  }
  __shared__ float rs[8], rss[8];
  __shared__ float mu_s, rstd_s;
#pragma unroll
  for (int o = 16; o > 0; o >>= 1) {
    s  += __shfl_down(s, o, 32);
    ss += __shfl_down(ss, o, 32);
  }
  if ((tid & 31) == 0) { rs[tid >> 5] = s; rss[tid >> 5] = ss; }
  __syncthreads();
  if (tid == 0) {
    float S = 0.f, SS = 0.f;
#pragma unroll
    for (int i = 0; i < 8; ++i) { S += rs[i]; SS += rss[i]; }
    float mu  = S * (1.f / 1024.f);
    float var = SS * (1.f / 1024.f) - mu * mu;
    mu_s = mu;
    rstd_s = rsqrtf(var + 1e-5f);
  }
  __syncthreads();
  const float mu = mu_s, rstd = rstd_s;
#pragma unroll
  for (int i = 0; i < 4; ++i) {
    int c = tid + i * 256;
    xn[(size_t)row * D_MODEL + c] = (_Float16)(((v[i] - mu) * rstd) * g[c] + b[c]);
  }
}

// ---------------------------------------------------------------------------
// Kernel 2: f32 -> f16 weight conversion (weights stay L2-resident)
// ---------------------------------------------------------------------------
__global__ void cvt_kernel(const float* __restrict__ src,
                           _Float16* __restrict__ dst, int n) {
  int i = blockIdx.x * 256 + threadIdx.x;
  if (i < n) dst[i] = (_Float16)src[i];
}

// ---------------------------------------------------------------------------
// Kernels 3 & 5: GEMM  C[M,N] = A[M,1024] * W[N,1024]^T + bias (+ residual)
// Block tile 128x128, 256 threads = 8 waves as 2x4 grid of 64x32 wave tiles.
// K staged through DOUBLE-BUFFERED LDS in 32-wide slabs via async-to-LDS;
// one barrier per slab; 8 x v_wmma_f32_16x16x32_f16 per slab per wave.
// MODE 0: f16 store (gi pre-activations).  MODE 1: f32 store + residual.
// ---------------------------------------------------------------------------
#define BM  128
#define BN  128
#define LDA 40   // padded LDS row stride (f16): 80B rows keep 16B alignment

template <int MODE>
__global__ __launch_bounds__(256) void gemm_kernel(
    const _Float16* __restrict__ A,     // [M, 1024]
    const _Float16* __restrict__ W,     // [N, 1024]
    const float* __restrict__ bias,     // [N]
    const float* __restrict__ residual, // [M, N] for MODE 1
    void* __restrict__ outp, int N) {
  __shared__ _Float16 As[2][BM * LDA];
  __shared__ _Float16 Ws[2][BN * LDA];
  const int tid = threadIdx.x;
  const int lane = tid & 31;
  const int wave = tid >> 5;
  const int laneHalf = lane >> 4;
  const int laneCol  = lane & 15;
  const int mblk = blockIdx.y * BM;
  const int nblk = blockIdx.x * BN;
  const int mwave = (wave >> 2) * 64;   // 2 wave-rows
  const int nwave = (wave & 3) * 32;    // 4 wave-cols

  v8f acc[4][2] = {};

  // cooperative staging coordinates: 2 threads per row, 16 f16 (2 x b128) each
  const int sr = tid >> 1, sh = (tid & 1) * 16;
  const _Float16* Ag = A + (size_t)(mblk + sr) * 1024 + sh;
  const _Float16* Wg = W + (size_t)(nblk + sr) * 1024 + sh;

  // prologue: stage slab 0 into buffer 0
  cp_b128(&As[0][sr * LDA + sh],     Ag);
  cp_b128(&As[0][sr * LDA + sh + 8], Ag + 8);
  cp_b128(&Ws[0][sr * LDA + sh],     Wg);
  cp_b128(&Ws[0][sr * LDA + sh + 8], Wg + 8);

  for (int k0 = 0; k0 < 1024; k0 += 32) {
    const int cur = (k0 >> 5) & 1;
    cp_wait();          // my staged slab has landed in LDS
    __syncthreads();    // everyone's slab landed; prev buffer fully consumed
    if (k0 + 32 < 1024) {
      const int nxt = cur ^ 1;
      cp_b128(&As[nxt][sr * LDA + sh],     Ag + k0 + 32);
      cp_b128(&As[nxt][sr * LDA + sh + 8], Ag + k0 + 40);
      cp_b128(&Ws[nxt][sr * LDA + sh],     Wg + k0 + 32);
      cp_b128(&Ws[nxt][sr * LDA + sh + 8], Wg + k0 + 40);
      __builtin_prefetch(Ag + k0 + 64, 0, 1);  // global_prefetch two slabs out
      __builtin_prefetch(Wg + k0 + 64, 0, 1);
    }

    v16h a[4], b[2];
#pragma unroll
    for (int mt = 0; mt < 4; ++mt)
      a[mt] = load_frag(As[cur], LDA, mwave + mt * 16 + laneCol, laneHalf, 0);
#pragma unroll
    for (int nt = 0; nt < 2; ++nt)
      b[nt] = load_frag(Ws[cur], LDA, nwave + nt * 16 + laneCol, laneHalf, 0);
#pragma unroll
    for (int mt = 0; mt < 4; ++mt)
#pragma unroll
      for (int nt = 0; nt < 2; ++nt)
        acc[mt][nt] = __builtin_amdgcn_wmma_f32_16x16x32_f16(
            false, a[mt], false, b[nt], (short)0, acc[mt][nt], false, false);
  }

  // epilogue: C/D layout (ISA 7.12.2): vgpr r -> M = r + 8*laneHalf, N = laneCol
#pragma unroll
  for (int mt = 0; mt < 4; ++mt)
#pragma unroll
    for (int nt = 0; nt < 2; ++nt) {
      const int tm = mblk + mwave + mt * 16;
      const int tn = nblk + nwave + nt * 16;
#pragma unroll
      for (int r = 0; r < 8; ++r) {
        const int row = tm + r + laneHalf * 8;
        const int col = tn + laneCol;
        float val = acc[mt][nt][r] + bias[col];
        if (MODE == 0) {
          ((_Float16*)outp)[(size_t)row * N + col] = (_Float16)val;
        } else {
          size_t idx = (size_t)row * N + col;
          ((float*)outp)[idx] = val + residual[idx];
        }
      }
    }
}

// ---------------------------------------------------------------------------
// Kernel 4: persistent cooperative GRU scan.
// 8 workgroups; workgroup g owns h-columns [g*128, g*128+128) => 384 W_hh gate
// rows (r/z/n). Each step: async-stage h (16x1024 f16) into LDS, each wave
// computes 3 WMMA n-tiles over K=1024 (W_hh f16 streamed from L2), gates
// exchanged via LDS, fused sigmoid/tanh epilogue, double-buffered h state,
// inter-WGP atomic-counter barrier (the recurrence critical path).
// ---------------------------------------------------------------------------
#define HPAD 1032  // 16B-aligned padded LDS stride for h
#define GPAD 392

__global__ __launch_bounds__(256) void gru_scan_kernel(
    const _Float16* __restrict__ gi,   // [NROWS, 3072] includes b_ih
    const _Float16* __restrict__ whh,  // [3072, 1024]
    const float* __restrict__ bhh,     // [3072]
    _Float16* __restrict__ hbuf,       // [2][16][1024], zero-initialized
    _Float16* __restrict__ y,          // [NROWS, 1024]
    unsigned int* __restrict__ ctr) {  // zero-initialized
  __shared__ _Float16 hA[16 * HPAD];
  __shared__ float ghs[16 * GPAD];
  const int tid = threadIdx.x;
  const int lane = tid & 31;
  const int wave = tid >> 5;
  const int laneHalf = lane >> 4;
  const int laneCol  = lane & 15;
  const int hbase = blockIdx.x * 128;
  const unsigned NWG = gridDim.x;  // 8

  for (int t = 0; t < SEQ; ++t) {
    const _Float16* hcur = hbuf + (size_t)(t & 1) * (16 * 1024);
    _Float16* hnext      = hbuf + (size_t)((t + 1) & 1) * (16 * 1024);

    // async-stage full h into LDS (64 f16 per thread, 8 x b128)
#pragma unroll
    for (int i = 0; i < 8; ++i) {
      int e = (tid + i * 256) * 8;          // 0..16376
      int m = e >> 10, c = e & 1023;
      cp_b128(&hA[m * HPAD + c], &hcur[e]);
    }
    cp_wait();
    __syncthreads();

    // gh = h @ W_hh_slice^T : 1 m-tile x 3 n-tiles per wave, K = 1024
    v8f acc[3] = {};
#pragma unroll 4
    for (int k0 = 0; k0 < 1024; k0 += 32) {
      v16h a = load_frag(hA, HPAD, laneCol, laneHalf, k0);
#pragma unroll
      for (int j = 0; j < 3; ++j) {
        const int tIdx = wave * 3 + j;                       // 0..23
        const int gate = tIdx >> 3, off = (tIdx & 7) * 16;
        const int rowb = gate * 1024 + hbase + off;          // W_hh gate row base
        v16h b = load_frag(whh, 1024, rowb + laneCol, laneHalf, k0);
        acc[j] = __builtin_amdgcn_wmma_f32_16x16x32_f16(
            false, a, false, b, (short)0, acc[j], false, false);
      }
    }
#pragma unroll
    for (int j = 0; j < 3; ++j) {
      const int coltile = (wave * 3 + j) * 16;               // 0..383
#pragma unroll
      for (int r = 0; r < 8; ++r)
        ghs[(r + laneHalf * 8) * GPAD + coltile + laneCol] = acc[j][r];
    }
    __syncthreads();

    // gate nonlinearities + state update for this block's 16x128 slice
#pragma unroll
    for (int i = 0; i < 8; ++i) {
      const int idx = tid + i * 256;      // 0..2047
      const int m = idx >> 7, cl = idx & 127;
      const int col = hbase + cl;
      const size_t girow = (size_t)(m * SEQ + t) * G3H;
      const float ir  = (float)gi[girow + col];
      const float iz  = (float)gi[girow + 1024 + col];
      const float in_ = (float)gi[girow + 2048 + col];
      const float hr = ghs[m * GPAD + cl]       + bhh[col];
      const float hz = ghs[m * GPAD + 128 + cl] + bhh[1024 + col];
      const float hn = ghs[m * GPAD + 256 + cl] + bhh[2048 + col];
      const float rg = 1.f / (1.f + __expf(-(ir + hr)));
      const float zg = 1.f / (1.f + __expf(-(iz + hz)));
      const float ng = tanhf(in_ + rg * hn);
      const float hold = (float)hA[m * HPAD + col];
      const float hnew = (1.f - zg) * ng + zg * hold;
      hnext[m * 1024 + col] = (_Float16)hnew;
      y[(size_t)(m * SEQ + t) * HGRU + col] = (_Float16)hnew;
    }

    // inter-workgroup barrier (monotonic counter, 8 WGs)
    __threadfence();
    if (tid == 0) {
      atomicAdd(ctr, 1u);
      const unsigned target = (unsigned)(t + 1) * NWG;
      while (__hip_atomic_load(ctr, __ATOMIC_RELAXED, __HIP_MEMORY_SCOPE_AGENT) < target)
        __builtin_amdgcn_s_sleep(2);
    }
    __syncthreads();
    __threadfence();
  }
}

// ---------------------------------------------------------------------------
// Kernel 6: h_last f16 -> f32 tail of d_out
// ---------------------------------------------------------------------------
__global__ void hlast_kernel(const _Float16* __restrict__ h, float* __restrict__ out) {
  int i = blockIdx.x * 256 + threadIdx.x;
  out[i] = (float)h[i];
}

// ---------------------------------------------------------------------------
// Workspace layout (bytes, 256-aligned), total ~334 MB
// ---------------------------------------------------------------------------
#define WS_XN    0ull
#define WS_GI    67108864ull    // 32768*1024*2
#define WS_Y     268435456ull   // + 32768*3072*2
#define WS_WIH   335544320ull   // + 32768*1024*2
#define WS_WHH   341835776ull   // + 3072*1024*2
#define WS_WOUT  348127232ull   // + 3072*1024*2
#define WS_HBUF  350224384ull   // + 1024*1024*2
#define WS_CTR   350289920ull   // + 2*16*1024*2

extern "C" void kernel_launch(void* const* d_in, const int* in_sizes, int n_in,
                              void* d_out, int out_size, void* d_ws, size_t ws_size,
                              hipStream_t stream) {
  const float* x    = (const float*)d_in[0];
  const float* lng  = (const float*)d_in[1];
  const float* lnb  = (const float*)d_in[2];
  const float* wih  = (const float*)d_in[3];
  const float* whh  = (const float*)d_in[4];
  const float* bih  = (const float*)d_in[5];
  const float* bhh  = (const float*)d_in[6];
  const float* wout = (const float*)d_in[7];
  const float* bout = (const float*)d_in[8];
  float* out = (float*)d_out;

  char* ws = (char*)d_ws;
  _Float16* xn     = (_Float16*)(ws + WS_XN);
  _Float16* gi     = (_Float16*)(ws + WS_GI);
  _Float16* yv     = (_Float16*)(ws + WS_Y);
  _Float16* wih16  = (_Float16*)(ws + WS_WIH);
  _Float16* whh16  = (_Float16*)(ws + WS_WHH);
  _Float16* wout16 = (_Float16*)(ws + WS_WOUT);
  _Float16* hbuf   = (_Float16*)(ws + WS_HBUF);
  unsigned int* ctr = (unsigned int*)(ws + WS_CTR);

  // zero h0 (both parity buffers) and the scan barrier counter every call
  (void)hipMemsetAsync(hbuf, 0, 2 * 16 * 1024 * sizeof(_Float16) + 256, stream);

  // 1) LayerNorm -> f16
  ln_kernel<<<NROWS, 256, 0, stream>>>(x, lng, lnb, xn);

  // 2) weights f32 -> f16 (L2-resident working set)
  cvt_kernel<<<(G3H * HGRU) / 256, 256, 0, stream>>>(wih, wih16, G3H * HGRU);
  cvt_kernel<<<(G3H * HGRU) / 256, 256, 0, stream>>>(whh, whh16, G3H * HGRU);
  cvt_kernel<<<(D_MODEL * HGRU) / 256, 256, 0, stream>>>(wout, wout16, D_MODEL * HGRU);

  // 3) gi = xn @ w_ih^T + b_ih   (32768 x 3072 x 1024, WMMA)
  gemm_kernel<0><<<dim3(G3H / BN, NROWS / BM), 256, 0, stream>>>(
      xn, wih16, bih, nullptr, gi, G3H);

  // 4) sequential GRU scan (persistent, 8 cooperative workgroups)
  gru_scan_kernel<<<8, 256, 0, stream>>>(gi, whh16, bhh, hbuf, yv, ctr);

  // 5) out = x + y @ w_out^T + b_out   (32768 x 1024 x 1024, WMMA, f32 out)
  gemm_kernel<1><<<dim3(D_MODEL / BN, NROWS / BM), 256, 0, stream>>>(
      yv, wout16, bout, x, out, D_MODEL);

  // 6) h_last tail (scan ends with h in parity-0 buffer: (2047+1)&1 == 0)
  hlast_kernel<<<64, 256, 0, stream>>>(hbuf, out + (size_t)NROWS * D_MODEL);
}
